// Model_63110249447980
// MI455X (gfx1250) — matrix operand order repaired and checked by reading, not compiled, not run on hardware
//
#include <hip/hip_runtime.h>
#include <hip/hip_bf16.h>

#define NSEQ 5
#define SEQ 512
#define NH 12
#define DHD 64
#define DMODEL 768
#define FFDIM 3072
#define NLAYER 4
#define NTOK (NSEQ*SEQ)          // 2560
#define MNUM 300
#define ENUM 100
#define TNUM 900
#define PNUM 400
#define EPADN 120
#define GRIDPIX (EPADN*EPADN)    // 14400
#define C1CH 256
#define K2DIM (9*256)            // 2304

typedef __attribute__((ext_vector_type(16))) __bf16 v16bf;
typedef __attribute__((ext_vector_type(8)))  float  v8f;

union FragA { v16bf v; unsigned u[8]; };
union AccF  { v8f   v; float    f[8]; };

#define HAS_ASYNC (__has_builtin(__builtin_amdgcn_global_load_async_to_lds_b128) && \
                   __has_builtin(__builtin_amdgcn_s_wait_asynccnt))

#if HAS_ASYNC
typedef int v4i_gcc __attribute__((vector_size(16)));
typedef __attribute__((address_space(1))) v4i_gcc* gptr_v4i;
typedef __attribute__((address_space(3))) v4i_gcc* lptr_v4i;
__device__ inline void async_cp16(const void* g, void* l) {
  __builtin_amdgcn_global_load_async_to_lds_b128(
      (gptr_v4i)g, (lptr_v4i)l, 0, 0);
}
#endif

__device__ inline unsigned short f2bf(float f) {
  union { float f; unsigned u; } x; x.f = f;
  unsigned u = x.u;
  unsigned r = u + 0x7FFFu + ((u >> 16) & 1u);
  return (unsigned short)(r >> 16);
}
__device__ inline float gelu_tanh(float x) {
  float x3 = x * x * x;
  return 0.5f * x * (1.0f + tanhf(0.7978845608028654f * (x + 0.044715f * x3)));
}

__device__ inline void block_reduce_2(float& a, float& b) {
  __shared__ float s1[256];
  __shared__ float s2[256];
  int t = threadIdx.x;
  s1[t] = a; s2[t] = b;
  __syncthreads();
  for (int o = 128; o > 0; o >>= 1) {
    if (t < o) { s1[t] += s1[t + o]; s2[t] += s2[t + o]; }
    __syncthreads();
  }
  a = s1[0]; b = s2[0];
  __syncthreads();
}

// ---------------------------------------------------------------------------
// bf16 WMMA GEMM, double-buffered LDS, async global->LDS A-tile staging.
// C[M,N] = A[M,K] * B[K,N] (+ epilogue).  A,B bf16; Cf fp32 / Cb bf16 (nullable).
// mode: 0=none, 1=+bias[n], 2=+bias[n] gelu, 3=+bias[n] relu, 4=+sbias[0]
// a_gather: 1 = implicit im2col over bf16 h1 (120x120, 256ch, 3x3 SAME)
// ---------------------------------------------------------------------------
__global__ __launch_bounds__(128) void wmma_gemm(
    const unsigned short* __restrict__ A, const unsigned short* __restrict__ B,
    const float* __restrict__ bias, const float* __restrict__ sbias,
    float* __restrict__ Cf, unsigned short* __restrict__ Cb,
    int M, int N, int K, int mode, int a_gather)
{
  __shared__ unsigned AsU[2][1024];   // 2 x (64 rows x 32 K) bf16
  __shared__ unsigned BsU[2][1024];   // 2 x ([n=64][k=32]) bf16 (transposed)

  int bm = blockIdx.x * 64;
  int bn = blockIdx.y * 64;
  int tid = threadIdx.x;
  int wave = tid >> 5;
  int lane = tid & 31;
  int l16 = lane & 15;
  int hl  = lane >> 4;

  bool fullA = (a_gather == 0) && (bm + 64 <= M) && ((K & 7) == 0);
  bool fullB = (bn + 64 <= N) && ((N & 7) == 0);

  auto stageA = [&](int kt, int bufi) {
    unsigned short* As = (unsigned short*)AsU[bufi];
    int k0 = kt * 32;
    if (fullA) {
#if HAS_ASYNC
      #pragma unroll
      for (int j = 0; j < 2; j++) {
        int c = tid + j * 128;
        int r = c >> 2, ko = (c & 3) * 8;
        async_cp16(A + (size_t)(bm + r) * K + k0 + ko, As + r * 32 + ko);
      }
#else
      #pragma unroll
      for (int j = 0; j < 2; j++) {
        int c = tid + j * 128;
        int r = c >> 2, ko = (c & 3) * 8;
        *(uint4*)(As + r * 32 + ko) = *(const uint4*)(A + (size_t)(bm + r) * K + k0 + ko);
      }
#endif
      if (k0 + 64 < K)
        __builtin_prefetch(A + (size_t)(bm + (tid >> 2)) * K + k0 + 64, 0, 1);
    } else {
      for (int i = tid; i < 2048; i += 128) {
        int r = i >> 5, cc = i & 31;
        int gr = bm + r;
        unsigned short v = 0;
        if (gr < M) {
          if (a_gather == 0) {
            v = A[(size_t)gr * K + k0 + cc];
          } else {
            int kg = k0 + cc;
            int tap = kg >> 8;          // ky*3+kx
            int cin = kg & 255;
            int ky = tap / 3, kx = tap % 3;
            int y = gr / EPADN + ky - 1;
            int x = gr % EPADN + kx - 1;
            if (y >= 0 && y < EPADN && x >= 0 && x < EPADN)
              v = A[((size_t)(y * EPADN + x)) * C1CH + cin];
          }
        }
        As[i] = v;
      }
    }
  };

  auto stageB = [&](int kt, int bufi) {
    unsigned short* Bs = (unsigned short*)BsU[bufi];
    int k0 = kt * 32;
    if (fullB) {
      #pragma unroll
      for (int j = 0; j < 2; j++) {
        int c = tid + j * 128;
        int k = c >> 3, n8 = (c & 7) * 8;
        uint4 t = *(const uint4*)(B + (size_t)(k0 + k) * N + bn + n8);
        const unsigned short* p16 = (const unsigned short*)&t;
        #pragma unroll
        for (int e = 0; e < 8; e++) Bs[(n8 + e) * 32 + k] = p16[e];
      }
      if (k0 + 64 < K)
        __builtin_prefetch(B + (size_t)(k0 + 32 + (tid >> 3)) * N + bn, 0, 1);
    } else {
      for (int i = tid; i < 2048; i += 128) {
        int k = i >> 6, n = i & 63;
        int gn = bn + n;
        Bs[n * 32 + k] = (gn < N) ? B[(size_t)(k0 + k) * N + gn] : (unsigned short)0;
      }
    }
  };

  AccF acc[4];
  #pragma unroll
  for (int t = 0; t < 4; t++)
    #pragma unroll
    for (int j = 0; j < 8; j++) acc[t].f[j] = 0.f;

  int nk = K >> 5;
  stageA(0, 0);
  stageB(0, 0);
#if HAS_ASYNC
  if (fullA) __builtin_amdgcn_s_wait_asynccnt(0);
#endif
  __syncthreads();

  for (int kt = 0; kt < nk; kt++) {
    int cur = kt & 1;
    if (kt + 1 < nk) { stageA(kt + 1, cur ^ 1); stageB(kt + 1, cur ^ 1); }

    const unsigned short* As = (const unsigned short*)AsU[cur];
    const unsigned short* Bs = (const unsigned short*)BsU[cur];

    FragA a;
    const unsigned* Au = (const unsigned*)&As[(wave * 16 + l16) * 32];
    #pragma unroll
    for (int vv = 0; vv < 8; vv++) {
      int kb = hl * 16 + 2 * vv;
      a.u[vv] = Au[kb >> 1];
    }
    #pragma unroll
    for (int nt = 0; nt < 4; nt++) {
      FragA b;
      const unsigned* Bu = (const unsigned*)&Bs[(nt * 16 + l16) * 32];
      #pragma unroll
      for (int vv = 0; vv < 8; vv++) {
        int kb = hl * 16 + 2 * vv;
        b.u[vv] = Bu[kb >> 1];
      }
      acc[nt].v = __builtin_amdgcn_wmma_f32_16x16x32_bf16(
          false, a.v, false, b.v, (short)0, acc[nt].v, false, false);
    }
#if HAS_ASYNC
    if (fullA) __builtin_amdgcn_s_wait_asynccnt(0);
#endif
    __syncthreads();
  }

  for (int nt = 0; nt < 4; nt++) {
    int gn = bn + nt * 16 + l16;
    if (gn >= N) continue;
    float bv = (mode >= 1 && mode <= 3) ? bias[gn] : 0.f;
    #pragma unroll
    for (int r = 0; r < 8; r++) {
      int gm = bm + wave * 16 + hl * 8 + r;
      if (gm >= M) continue;
      float x = acc[nt].f[r];
      if (mode == 1) x += bv;
      else if (mode == 2) x = gelu_tanh(x + bv);
      else if (mode == 3) x = fmaxf(x + bv, 0.f);
      else if (mode == 4) x += sbias[0];
      if (Cf) Cf[(size_t)gm * N + gn] = x;
      if (Cb) Cb[(size_t)gm * N + gn] = f2bf(x);
    }
  }
}

// ---------------------------------------------------------------------------
// Fused attention: one wave per (b, h, 16-query tile).  Q/K/V bf16 in global.
// ---------------------------------------------------------------------------
__global__ __launch_bounds__(32) void attn_kernel(
    const unsigned short* __restrict__ qg, const unsigned short* __restrict__ kg,
    const unsigned short* __restrict__ vg, const float* __restrict__ mask,
    unsigned short* __restrict__ ctxb)
{
  __shared__ float sc[16 * SEQ];          // 32 KB
  __shared__ unsigned prU[16 * SEQ / 2];  // 16 KB bf16 probs
  unsigned short* pr = (unsigned short*)prU;

  int id = blockIdx.x;
  int qt = id & 31; id >>= 5;
  int h = id % NH;
  int b = id / NH;
  int q0 = qt * 16;

  int lane = threadIdx.x;
  int l16 = lane & 15;
  int hl = lane >> 4;

  FragA qa[2];
  {
    const unsigned* Qu = (const unsigned*)(qg + ((size_t)(b * SEQ + q0 + l16)) * DMODEL + h * DHD);
    #pragma unroll
    for (int d = 0; d < 2; d++) {
      #pragma unroll
      for (int vv = 0; vv < 8; vv++) {
        int kof = hl * 16 + 2 * vv;
        qa[d].u[vv] = Qu[(d * 32 + kof) >> 1];
      }
    }
  }
  for (int kt = 0; kt < SEQ / 16; kt++) {
    AccF accs;
    #pragma unroll
    for (int j = 0; j < 8; j++) accs.f[j] = 0.f;
    #pragma unroll
    for (int d = 0; d < 2; d++) {
      FragA bfr;  // B = K^T : B[dh][key] = K[key][dh]
      const unsigned* Ku = (const unsigned*)(kg + ((size_t)(b * SEQ + kt * 16 + l16)) * DMODEL + h * DHD + d * 32);
      #pragma unroll
      for (int vv = 0; vv < 8; vv++) {
        int kof = hl * 16 + 2 * vv;
        bfr.u[vv] = Ku[kof >> 1];
      }
      accs.v = __builtin_amdgcn_wmma_f32_16x16x32_bf16(
          false, qa[d].v, false, bfr.v, (short)0, accs.v, false, false);
    }
    int key = kt * 16 + l16;
    float biasv = (1.0f - mask[b * SEQ + key]) * -1e9f;
    #pragma unroll
    for (int r = 0; r < 8; r++)
      sc[(hl * 8 + r) * SEQ + key] = accs.f[r] * 0.125f + biasv;
  }
  __syncthreads();
  {
    int row = l16;
    int kbase = hl * 256;
    float mx = -1e30f;
    for (int kk = 0; kk < 256; kk++) mx = fmaxf(mx, sc[row * SEQ + kbase + kk]);
    mx = fmaxf(mx, __shfl_xor(mx, 16));
    float sum = 0.f;
    for (int kk = 0; kk < 256; kk++) {
      float e = __expf(sc[row * SEQ + kbase + kk] - mx);
      sc[row * SEQ + kbase + kk] = e;
      sum += e;
    }
    sum += __shfl_xor(sum, 16);
    float inv = 1.0f / sum;
    for (int kk = 0; kk < 256; kk++)
      pr[row * SEQ + kbase + kk] = f2bf(sc[row * SEQ + kbase + kk] * inv);
  }
  __syncthreads();
  for (int nt = 0; nt < 4; nt++) {
    AccF acc;
    #pragma unroll
    for (int j = 0; j < 8; j++) acc.f[j] = 0.f;
    for (int kc = 0; kc < SEQ / 32; kc++) {
      FragA a, bfr;
      const unsigned* Pu = (const unsigned*)&pr[l16 * SEQ + kc * 32];
      #pragma unroll
      for (int vv = 0; vv < 8; vv++) {
        int kof = hl * 16 + 2 * vv;
        a.u[vv] = Pu[kof >> 1];
      }
      #pragma unroll
      for (int vv = 0; vv < 8; vv++) {
        int kof = hl * 16 + 2 * vv;
        unsigned short v0 = vg[((size_t)(b * SEQ + kc * 32 + kof)) * DMODEL + h * DHD + nt * 16 + l16];
        unsigned short v1 = vg[((size_t)(b * SEQ + kc * 32 + kof + 1)) * DMODEL + h * DHD + nt * 16 + l16];
        bfr.u[vv] = (unsigned)v0 | ((unsigned)v1 << 16);
      }
      acc.v = __builtin_amdgcn_wmma_f32_16x16x32_bf16(
          false, a.v, false, bfr.v, (short)0, acc.v, false, false);
    }
    #pragma unroll
    for (int r = 0; r < 8; r++)
      ctxb[((size_t)(b * SEQ + q0 + hl * 8 + r)) * DMODEL + h * DHD + nt * 16 + l16] = f2bf(acc.f[r]);
  }
}

// ---------------------------------------------------------------------------
// LayerNorm kernels (row = block, 256 threads, D=768); emit fp32 + bf16
// ---------------------------------------------------------------------------
__global__ __launch_bounds__(256) void embed_ln_kernel(
    const int* __restrict__ ids, const float* __restrict__ emb, const float* __restrict__ pos,
    const float* __restrict__ g, const float* __restrict__ b,
    float* __restrict__ out, unsigned short* __restrict__ outb)
{
  int row = blockIdx.x;
  int s = row & (SEQ - 1);
  int id = ids[row];
  float v[3], sum = 0.f, sq = 0.f;
  for (int j = 0; j < 3; j++) {
    int d = threadIdx.x + j * 256;
    float x = emb[(size_t)id * DMODEL + d] + pos[(size_t)s * DMODEL + d];
    v[j] = x; sum += x; sq += x * x;
  }
  block_reduce_2(sum, sq);
  float mean = sum * (1.0f / DMODEL);
  float var = sq * (1.0f / DMODEL) - mean * mean;
  float inv = rsqrtf(var + 1e-12f);
  for (int j = 0; j < 3; j++) {
    int d = threadIdx.x + j * 256;
    float y = (v[j] - mean) * inv * g[d] + b[d];
    out[(size_t)row * DMODEL + d] = y;
    outb[(size_t)row * DMODEL + d] = f2bf(y);
  }
}

__global__ __launch_bounds__(256) void add_ln_kernel(
    const float* __restrict__ xa, const float* __restrict__ xr,
    const float* __restrict__ g, const float* __restrict__ b,
    float* __restrict__ out, unsigned short* __restrict__ outb)
{
  int row = blockIdx.x;
  float v[3], sum = 0.f, sq = 0.f;
  for (int j = 0; j < 3; j++) {
    int d = threadIdx.x + j * 256;
    float x = xa[(size_t)row * DMODEL + d] + xr[(size_t)row * DMODEL + d];
    v[j] = x; sum += x; sq += x * x;
  }
  block_reduce_2(sum, sq);
  float mean = sum * (1.0f / DMODEL);
  float var = sq * (1.0f / DMODEL) - mean * mean;
  float inv = rsqrtf(var + 1e-12f);
  for (int j = 0; j < 3; j++) {
    int d = threadIdx.x + j * 256;
    float y = (v[j] - mean) * inv * g[d] + b[d];
    out[(size_t)row * DMODEL + d] = y;
    outb[(size_t)row * DMODEL + d] = f2bf(y);
  }
}

// ---------------------------------------------------------------------------
__global__ __launch_bounds__(256) void cvt_kernel(
    const float* __restrict__ in, unsigned short* __restrict__ out, int n)
{
  int i = blockIdx.x * 256 + threadIdx.x;
  if (i < n) out[i] = f2bf(in[i]);
}

// ---------------------------------------------------------------------------
// Pooling (deterministic scans, no float atomics)
// ---------------------------------------------------------------------------
__global__ __launch_bounds__(256) void mention_kernel(
    const float* __restrict__ tokens, const int* __restrict__ span_idx,
    const int* __restrict__ mids, float* __restrict__ m_emb)
{
  int m = blockIdx.x;
  float cnt = 0.f;
  for (int t = 0; t < TNUM; t++) cnt += (mids[t] == m) ? 1.f : 0.f;
  float inv = 1.0f / cnt;
  for (int d = threadIdx.x; d < DMODEL; d += 256) {
    float s = 0.f;
    for (int t = 0; t < TNUM; t++)
      if (mids[t] == m) s += tokens[(size_t)span_idx[t] * DMODEL + d];
    m_emb[(size_t)m * DMODEL + d] = s * inv;
  }
}

__global__ __launch_bounds__(256) void entity_kernel(
    const float* __restrict__ m_emb, const int* __restrict__ eom, float* __restrict__ ent)
{
  int e = blockIdx.x;
  float cnt = 0.f; int last = 0;
  for (int mm = 0; mm < MNUM; mm++)
    if (eom[mm] == e) { cnt += 1.f; last = mm; }
  float inv = 1.0f / cnt;
  for (int d = threadIdx.x; d < DMODEL; d += 256) {
    float s = 0.f;
    for (int mm = 0; mm < MNUM; mm++)
      if (eom[mm] == e) s += m_emb[(size_t)mm * DMODEL + d];
    ent[(size_t)e * DMODEL + d] = (s + m_emb[(size_t)last * DMODEL + d]) * inv;
  }
}

__global__ __launch_bounds__(256) void ep_build_kernel(
    const float* __restrict__ ent, float* __restrict__ ep, unsigned short* __restrict__ epb)
{
  int idx = blockIdx.x * 256 + threadIdx.x;
  if (idx >= EPADN * DMODEL) return;
  float v = (idx < ENUM * DMODEL) ? ent[idx] : 0.f;
  ep[idx] = v;
  epb[idx] = f2bf(v);
}

__global__ __launch_bounds__(256) void epn_kernel(
    const float* __restrict__ ep, float* __restrict__ epn, unsigned short* __restrict__ epnb)
{
  int r = blockIdx.x;
  float s = 0.f, dummy = 0.f;
  float v[3];
  for (int j = 0; j < 3; j++) {
    int d = threadIdx.x + j * 256;
    float x = ep[(size_t)r * DMODEL + d];
    v[j] = x; s += x * x;
  }
  block_reduce_2(s, dummy);
  float inv = 1.0f / (sqrtf(s) + 1e-13f);
  for (int j = 0; j < 3; j++) {
    int d = threadIdx.x + j * 256;
    float y = v[j] * inv;
    epn[(size_t)r * DMODEL + d] = y;
    epnb[(size_t)r * DMODEL + d] = f2bf(y);
  }
}

__global__ __launch_bounds__(256) void transpose_cvt_kernel(
    const float* __restrict__ in, unsigned short* __restrict__ outp, int R, int C)
{
  int idx = blockIdx.x * 256 + threadIdx.x;
  if (idx >= R * C) return;
  int r = idx / C, c = idx % C;
  outp[(size_t)c * R + r] = f2bf(in[idx]);
}

// ---------------------------------------------------------------------------
// Convolutions
// ---------------------------------------------------------------------------
__global__ __launch_bounds__(256) void conv1_kernel(
    const float* __restrict__ c0, const float* __restrict__ c1, const float* __restrict__ c2,
    const float* __restrict__ w, const float* __restrict__ bias, unsigned short* __restrict__ out)
{
  int pix = blockIdx.x;
  int oc = threadIdx.x;
  int y = pix / EPADN, x = pix % EPADN;
  float acc = bias[oc];
  const float* chans[3] = { c0, c1, c2 };
  for (int c = 0; c < 3; c++)
    for (int ky = 0; ky < 3; ky++)
      for (int kx = 0; kx < 3; kx++) {
        int yy = y + ky - 1, xx = x + kx - 1;
        if (yy >= 0 && yy < EPADN && xx >= 0 && xx < EPADN)
          acc += chans[c][yy * EPADN + xx] * w[((oc * 3 + c) * 3 + ky) * 3 + kx];
      }
  out[(size_t)pix * C1CH + oc] = f2bf(fmaxf(acc, 0.f));
}

// conv2 weight repack OIHW -> bf16 [K=tap*256+cin][N=oc]
__global__ __launch_bounds__(256) void conv2_repack(const float* __restrict__ w, unsigned short* __restrict__ B2) {
  int kidx = blockIdx.x;     // 0..2303
  int o = threadIdx.x;       // 0..255
  int tap = kidx >> 8;
  int cin = kidx & 255;
  int ky = tap / 3, kx = tap % 3;
  B2[(size_t)kidx * C1CH + o] = f2bf(w[(size_t)(o * 256 + cin) * 9 + ky * 3 + kx]);
}

// ---------------------------------------------------------------------------
// Pair head
// ---------------------------------------------------------------------------
__global__ __launch_bounds__(256) void ht_gather_kernel(
    const unsigned short* __restrict__ amap, const int* __restrict__ htp, unsigned short* __restrict__ ht0)
{
  int p = blockIdx.x;
  int c = threadIdx.x;
  int y = htp[p * 2 + 0];
  int x = htp[p * 2 + 1];
  ht0[(size_t)p * C1CH + c] = amap[((size_t)(y * EPADN + x)) * C1CH + c];
}

__global__ __launch_bounds__(256) void final_kernel(
    const float* __restrict__ ent, const float* __restrict__ ht1,
    const int* __restrict__ htp, const float* __restrict__ Wcls,
    const float* __restrict__ bcls, float* __restrict__ out)
{
  __shared__ float hs[832];
  __shared__ float ts[832];
  __shared__ float red[5][256];
  int p = blockIdx.x;
  int tid = threadIdx.x;
  int hi = htp[p * 2 + 0];
  int ti = htp[p * 2 + 1];
  for (int i = tid; i < 832; i += 256) {
    float hv = (i < 768) ? ent[(size_t)hi * DMODEL + i] : ht1[p * 64 + (i - 768)];
    float tv = (i < 768) ? ent[(size_t)ti * DMODEL + i] : ht1[p * 64 + (i - 768)];
    hs[i] = tanhf(hv);
    ts[i] = tanhf(tv);
  }
  __syncthreads();
  float acc[5] = {0.f, 0.f, 0.f, 0.f, 0.f};
  for (int idx = tid; idx < 832; idx += 256) {
    int i = idx >> 6;
    float hv = hs[idx];
    const float* wrow = Wcls + (size_t)idx * 64 * 5;
    for (int kk = 0; kk < 64; kk++) {
      float hvt = hv * ts[(i << 6) + kk];
      #pragma unroll
      for (int c = 0; c < 5; c++) acc[c] += hvt * wrow[kk * 5 + c];
    }
  }
  #pragma unroll
  for (int c = 0; c < 5; c++) red[c][tid] = acc[c];
  __syncthreads();
  for (int o = 128; o > 0; o >>= 1) {
    if (tid < o) {
      #pragma unroll
      for (int c = 0; c < 5; c++) red[c][tid] += red[c][tid + o];
    }
    __syncthreads();
  }
  if (tid < 5) out[p * 5 + tid] = red[tid][0] + bcls[tid];
}

// ---------------------------------------------------------------------------
extern "C" void kernel_launch(void* const* d_in, const int* in_sizes, int n_in,
                              void* d_out, int out_size, void* d_ws, size_t ws_size,
                              hipStream_t stream) {
  (void)in_sizes; (void)n_in; (void)out_size; (void)ws_size;
  const int*   ids      = (const int*)  d_in[0];
  const float* masks    = (const float*)d_in[1];
  const int*   span_idx = (const int*)  d_in[2];
  const int*   mids     = (const int*)  d_in[3];
  const int*   eom      = (const int*)  d_in[4];
  const int*   htp      = (const int*)  d_in[5];
  const float* emb      = (const float*)d_in[6];
  const float* pos      = (const float*)d_in[7];
  const float* lng      = (const float*)d_in[8];
  const float* lnb      = (const float*)d_in[9];
  const float* Wq   = (const float*)d_in[10];
  const float* Wk   = (const float*)d_in[11];
  const float* Wv   = (const float*)d_in[12];
  const float* Wo   = (const float*)d_in[13];
  const float* bq   = (const float*)d_in[14];
  const float* bk   = (const float*)d_in[15];
  const float* bv   = (const float*)d_in[16];
  const float* bo   = (const float*)d_in[17];
  const float* ln1g = (const float*)d_in[18];
  const float* ln1b = (const float*)d_in[19];
  const float* Wff1 = (const float*)d_in[20];
  const float* bff1 = (const float*)d_in[21];
  const float* Wff2 = (const float*)d_in[22];
  const float* bff2 = (const float*)d_in[23];
  const float* ln2g = (const float*)d_in[24];
  const float* ln2b = (const float*)d_in[25];
  const float* c1w  = (const float*)d_in[26];
  const float* c1b  = (const float*)d_in[27];
  const float* c2w  = (const float*)d_in[28];
  const float* c2b  = (const float*)d_in[29];
  const float* Wbil = (const float*)d_in[30];
  const float* bbil = (const float*)d_in[31];
  const float* Wue  = (const float*)d_in[32];
  const float* bue  = (const float*)d_in[33];
  const float* Wcls = (const float*)d_in[34];
  const float* bcls = (const float*)d_in[35];
  float* out = (float*)d_out;

  char* base = (char*)d_ws;
  size_t off = 0;
  auto alloc_f = [&](size_t n) -> float* {
    float* p = (float*)(base + off);
    off = (off + n * sizeof(float) + 255) & ~(size_t)255;
    return p;
  };
  auto alloc_b = [&](size_t n) -> unsigned short* {
    unsigned short* p = (unsigned short*)(base + off);
    off = (off + n * sizeof(unsigned short) + 255) & ~(size_t)255;
    return p;
  };
  float* x0   = alloc_f((size_t)NTOK * DMODEL);
  float* x1   = alloc_f((size_t)NTOK * DMODEL);
  float* tmp  = alloc_f((size_t)NTOK * DMODEL);
  unsigned short* x0b  = alloc_b((size_t)NTOK * DMODEL);
  unsigned short* x1b  = alloc_b((size_t)NTOK * DMODEL);
  unsigned short* qb   = alloc_b((size_t)NTOK * DMODEL);
  unsigned short* kb   = alloc_b((size_t)NTOK * DMODEL);
  unsigned short* vb   = alloc_b((size_t)NTOK * DMODEL);
  unsigned short* ctxb = alloc_b((size_t)NTOK * DMODEL);
  unsigned short* ffb  = alloc_b((size_t)NTOK * FFDIM);
  float* m_emb = alloc_f((size_t)MNUM * DMODEL);
  float* ent  = alloc_f((size_t)ENUM * DMODEL);
  float* ep   = alloc_f((size_t)EPADN * DMODEL);
  float* epn  = alloc_f((size_t)EPADN * DMODEL);
  unsigned short* epb   = alloc_b((size_t)EPADN * DMODEL);
  unsigned short* epnb  = alloc_b((size_t)EPADN * DMODEL);
  unsigned short* epTb  = alloc_b((size_t)DMODEL * EPADN);
  unsigned short* epnTb = alloc_b((size_t)DMODEL * EPADN);
  unsigned short* t1b   = alloc_b((size_t)EPADN * DMODEL);
  float* dotb = alloc_f((size_t)GRIDPIX);
  float* cosb = alloc_f((size_t)GRIDPIX);
  float* bilb = alloc_f((size_t)GRIDPIX);
  unsigned short* h1b   = alloc_b((size_t)GRIDPIX * C1CH);
  unsigned short* amapb = alloc_b((size_t)GRIDPIX * C1CH);
  unsigned short* B2b   = alloc_b((size_t)K2DIM * C1CH);
  unsigned short* ht0b  = alloc_b((size_t)PNUM * C1CH);
  float* ht1f = alloc_f((size_t)PNUM * 64);
  // bf16 weights
  unsigned short* Wqb = alloc_b((size_t)NLAYER * DMODEL * DMODEL);
  unsigned short* Wkb = alloc_b((size_t)NLAYER * DMODEL * DMODEL);
  unsigned short* Wvb = alloc_b((size_t)NLAYER * DMODEL * DMODEL);
  unsigned short* Wob = alloc_b((size_t)NLAYER * DMODEL * DMODEL);
  unsigned short* Wff1b = alloc_b((size_t)NLAYER * DMODEL * FFDIM);
  unsigned short* Wff2b = alloc_b((size_t)NLAYER * DMODEL * FFDIM);
  unsigned short* Wbilb = alloc_b((size_t)DMODEL * DMODEL);
  unsigned short* Wueb  = alloc_b((size_t)C1CH * 64);

  // ---- weight conversions (bf16-resident working set; fits in L2) ----
  int DD4 = NLAYER * DMODEL * DMODEL;
  int FF4 = NLAYER * DMODEL * FFDIM;
  cvt_kernel<<<(DD4 + 255) / 256, 256, 0, stream>>>(Wq, Wqb, DD4);
  cvt_kernel<<<(DD4 + 255) / 256, 256, 0, stream>>>(Wk, Wkb, DD4);
  cvt_kernel<<<(DD4 + 255) / 256, 256, 0, stream>>>(Wv, Wvb, DD4);
  cvt_kernel<<<(DD4 + 255) / 256, 256, 0, stream>>>(Wo, Wob, DD4);
  cvt_kernel<<<(FF4 + 255) / 256, 256, 0, stream>>>(Wff1, Wff1b, FF4);
  cvt_kernel<<<(FF4 + 255) / 256, 256, 0, stream>>>(Wff2, Wff2b, FF4);
  cvt_kernel<<<(DMODEL * DMODEL + 255) / 256, 256, 0, stream>>>(Wbil, Wbilb, DMODEL * DMODEL);
  cvt_kernel<<<(C1CH * 64 + 255) / 256, 256, 0, stream>>>(Wue, Wueb, C1CH * 64);

  embed_ln_kernel<<<NTOK, 256, 0, stream>>>(ids, emb, pos, lng, lnb, x0, x0b);

  dim3 g768(NTOK / 64, DMODEL / 64);   // 40 x 12
  dim3 gff(NTOK / 64, FFDIM / 64);     // 40 x 48
  for (int l = 0; l < NLAYER; l++) {
    const unsigned short* wq = Wqb + (size_t)l * DMODEL * DMODEL;
    const unsigned short* wk = Wkb + (size_t)l * DMODEL * DMODEL;
    const unsigned short* wv = Wvb + (size_t)l * DMODEL * DMODEL;
    const unsigned short* wo = Wob + (size_t)l * DMODEL * DMODEL;
    wmma_gemm<<<g768, 128, 0, stream>>>(x0b, wq, bq + l * DMODEL, nullptr, nullptr, qb, NTOK, DMODEL, DMODEL, 1, 0);
    wmma_gemm<<<g768, 128, 0, stream>>>(x0b, wk, bk + l * DMODEL, nullptr, nullptr, kb, NTOK, DMODEL, DMODEL, 1, 0);
    wmma_gemm<<<g768, 128, 0, stream>>>(x0b, wv, bv + l * DMODEL, nullptr, nullptr, vb, NTOK, DMODEL, DMODEL, 1, 0);
    attn_kernel<<<NSEQ * NH * (SEQ / 16), 32, 0, stream>>>(qb, kb, vb, masks, ctxb);
    wmma_gemm<<<g768, 128, 0, stream>>>(ctxb, wo, bo + l * DMODEL, nullptr, tmp, nullptr, NTOK, DMODEL, DMODEL, 1, 0);
    add_ln_kernel<<<NTOK, 256, 0, stream>>>(x0, tmp, ln1g + l * DMODEL, ln1b + l * DMODEL, x1, x1b);
    wmma_gemm<<<gff, 128, 0, stream>>>(x1b, Wff1b + (size_t)l * DMODEL * FFDIM, bff1 + l * FFDIM, nullptr,
                                       nullptr, ffb, NTOK, FFDIM, DMODEL, 2, 0);
    wmma_gemm<<<g768, 128, 0, stream>>>(ffb, Wff2b + (size_t)l * FFDIM * DMODEL, bff2 + l * DMODEL, nullptr,
                                        tmp, nullptr, NTOK, DMODEL, FFDIM, 1, 0);
    add_ln_kernel<<<NTOK, 256, 0, stream>>>(x1, tmp, ln2g + l * DMODEL, ln2b + l * DMODEL, x0, x0b);
  }

  mention_kernel<<<MNUM, 256, 0, stream>>>(x0, span_idx, mids, m_emb);
  entity_kernel<<<ENUM, 256, 0, stream>>>(m_emb, eom, ent);
  ep_build_kernel<<<(EPADN * DMODEL) / 256, 256, 0, stream>>>(ent, ep, epb);
  epn_kernel<<<EPADN, 256, 0, stream>>>(ep, epn, epnb);
  transpose_cvt_kernel<<<(EPADN * DMODEL) / 256, 256, 0, stream>>>(ep, epTb, EPADN, DMODEL);
  transpose_cvt_kernel<<<(EPADN * DMODEL) / 256, 256, 0, stream>>>(epn, epnTb, EPADN, DMODEL);

  dim3 gEE(2, 2);
  dim3 gED(2, 12);
  wmma_gemm<<<gEE, 128, 0, stream>>>(epb, epTb, nullptr, nullptr, dotb, nullptr, EPADN, EPADN, DMODEL, 0, 0);
  wmma_gemm<<<gEE, 128, 0, stream>>>(epnb, epnTb, nullptr, nullptr, cosb, nullptr, EPADN, EPADN, DMODEL, 0, 0);
  wmma_gemm<<<gED, 128, 0, stream>>>(epb, Wbilb, nullptr, nullptr, nullptr, t1b, EPADN, DMODEL, DMODEL, 0, 0);
  wmma_gemm<<<gEE, 128, 0, stream>>>(t1b, epTb, nullptr, bbil, bilb, nullptr, EPADN, EPADN, DMODEL, 4, 0);

  conv1_kernel<<<GRIDPIX, 256, 0, stream>>>(dotb, cosb, bilb, c1w, c1b, h1b);
  conv2_repack<<<K2DIM, 256, 0, stream>>>(c2w, B2b);
  dim3 gc2(GRIDPIX / 64, C1CH / 64);   // 225 x 4
  wmma_gemm<<<gc2, 128, 0, stream>>>(h1b, B2b, c2b, nullptr, nullptr, amapb, GRIDPIX, C1CH, K2DIM, 3, 1);

  ht_gather_kernel<<<PNUM, 256, 0, stream>>>(amapb, htp, ht0b);
  dim3 ght((PNUM + 63) / 64, 1);
  wmma_gemm<<<ght, 128, 0, stream>>>(ht0b, Wueb, bue, nullptr, ht1f, nullptr, PNUM, 64, C1CH, 1, 0);
  final_kernel<<<PNUM, 256, 0, stream>>>(ent, ht1f, htp, Wcls, bcls, out);
}
